// DrugDiseaseHeteroGAT_84834194031306
// MI455X (gfx1250) — compile-verified, matrix operand order checked by reading, not run on hardware
//
#include <hip/hip_runtime.h>
#include <hip/hip_bf16.h>

// ---------------------------------------------------------------------------
// DrugDiseaseHeteroGAT on MI455X (gfx1250, wave32, WMMA + async-to-LDS)
// - Feature matrices converted to bf16 once per layer (ReLU fused), since each
//   is consumed by 14 GEMMs/layer. GEMM hot loop has zero conversion VALU.
// - GEMM: 64-row tiles, 4 acc/wave, A bf16 tiles double-buffered in LDS via
//   global_load_async_to_lds_b128 (+ s_wait_asynccnt), B prepacked to exact
//   WMMA fragment order (L2-resident), 4x v_wmma_f32_16x16x32_bf16 per K-step.
// - Edge softmax/scatter: wave-per-edge float4 gathers + L2 atomics
//   (~1.1 GB/layer -> the real roofline term at 23.3 TB/s).
// ---------------------------------------------------------------------------

typedef __attribute__((ext_vector_type(16))) __bf16 v16bf;
typedef __attribute__((ext_vector_type(8)))  float  v8f;
typedef __attribute__((ext_vector_type(4)))  float  f32x4;

#define NDRUG 20000
#define NDIS  40000
#define DIN   256
#define HIDC  128
#define NEDGE 150000

__device__ __forceinline__ __bf16 f2bf(float f) {
  unsigned u = __float_as_uint(f);
  u += 0x7FFFu + ((u >> 16) & 1u);          // round-to-nearest-even
  unsigned short s = (unsigned short)(u >> 16);
  __bf16 r;
  __builtin_memcpy(&r, &s, 2);
  return r;
}

// order-preserving float <-> uint mapping (for atomicMax-based segment max)
__device__ __forceinline__ unsigned fenc(float f) {
  unsigned b = __float_as_uint(f);
  return (b & 0x80000000u) ? ~b : (b | 0x80000000u);
}
__device__ __forceinline__ float fdec(unsigned e) {
  return __uint_as_float((e & 0x80000000u) ? (e & 0x7FFFFFFFu) : ~e);
}
#define ENC_NEG_INF 0x007FFFFFu  // fenc(-inf)

// f32 [n] -> bf16 [n], optional fused ReLU; 8 elems/thread, 16B stores
__global__ __launch_bounds__(256)
void cvt_bf16(const float* __restrict__ X, __bf16* __restrict__ Y, size_t n,
              int relu) {
  size_t i = ((size_t)blockIdx.x * 256 + threadIdx.x) * 8;
  if (i >= n) return;
  f32x4 a = *(const f32x4*)(X + i);
  f32x4 b = *(const f32x4*)(X + i + 4);
  float t[8];
  *(f32x4*)(t + 0) = a;
  *(f32x4*)(t + 4) = b;
  __bf16 o[8];
#pragma unroll
  for (int e = 0; e < 8; ++e) {
    float x = t[e];
    if (relu) x = fmaxf(x, 0.0f);
    o[e] = f2bf(x);
  }
  *(f32x4*)(Y + i) = *(f32x4*)o;
}

// ---------------------------------------------------------------------------
// Pre-pack B[K,128] (f32) into bf16 WMMA B-fragment order:
//   Bp[(kt*128 + n)*32 + k] = bf16(B[(kt*32 + k)*128 + n])
// Lane (wave w, lane l, half hi) reads its 16 elements as two 16-byte runs:
// k in {8hi..8hi+7} and {16+8hi..23+8hi} of column n = 16w + (l&15).
// ---------------------------------------------------------------------------
__global__ __launch_bounds__(256)
void pack_B(const float* __restrict__ B, __bf16* __restrict__ Bp, int K) {
  int i = blockIdx.x * 256 + threadIdx.x;     // over K*128
  if (i >= K * 128) return;
  int kt = i >> 12;                            // / (32*128)
  int r  = i & 4095;
  int k  = r >> 7;
  int n  = r & 127;                            // n fastest -> coalesced reads
  Bp[((size_t)kt * 128 + n) * 32 + k] = f2bf(B[(size_t)(kt * 32 + k) * 128 + n]);
}

// ---------------------------------------------------------------------------
// C[M,128] = act( A[M,K] @ B[K,128] + bias ), A and Bpack already bf16.
// Block: 256 threads (8 waves). M-tile = 64 rows (4 sub-tiles of 16/wave),
// wave w owns columns [16w, 16w+16). A tiles (64x32 bf16 = 4KB) double
// buffered in LDS via async-to-LDS; B fragments straight from L2.
// ---------------------------------------------------------------------------
template <bool HAS_BIAS, bool RELU_OUT>
__global__ __launch_bounds__(256)
void gemm_wmma(const __bf16* __restrict__ A, const __bf16* __restrict__ Bpack,
               const float* __restrict__ bias, float* __restrict__ C,
               int M, int K) {
  __shared__ __attribute__((aligned(16))) __bf16 Abf[2][64 * 32];  // 2 x 4KB

  const int tid  = threadIdx.x;
  const int wave = tid >> 5;
  const int lane = tid & 31;
  const int hi   = lane >> 4;     // half-wave select (ISA 7.12.2 operand layout)
  const int rc   = lane & 15;     // row-in-subtile (A) / column (B,C)
  const int m0   = blockIdx.x * 64;

  // branch-free async staging of one 64x32 bf16 A tile (256 x 16B transfers)
  auto stage = [&](int buf, int k0) {
    int idx = tid * 8;                          // 8 bf16 = 16B per thread
    int m = idx >> 5, k = idx & 31;
    int mr = (m0 + m < M) ? (m0 + m) : (M - 1); // clamp: no divergence
    unsigned long long ga =
        (unsigned long long)(const void*)(A + (size_t)mr * K + k0 + k);
    unsigned lds = (unsigned)(unsigned long long)(const void*)&Abf[buf][idx];
    asm volatile("global_load_async_to_lds_b128 %0, %1, off"
                 :: "v"(lds), "v"(ga) : "memory");
  };

  v8f acc[4] = {v8f{}, v8f{}, v8f{}, v8f{}};

  const int ktiles = K >> 5;
  stage(0, 0);
  asm volatile("s_wait_asynccnt 0x0" ::: "memory");
  __syncthreads();

  for (int kt = 0; kt < ktiles; ++kt) {
    if (kt + 1 < ktiles) stage((kt + 1) & 1, (kt + 1) << 5);  // overlap copy

    // B fragment: two 16B global loads from prepacked bf16 (shared by 4 WMMAs)
    v16bf b;
    {
      const __bf16* bp =
          Bpack + (((size_t)kt << 7) + (wave << 4) + rc) * 32 + (hi << 3);
      f32x4 blo = *(const f32x4*)bp;
      f32x4 bhi = *(const f32x4*)(bp + 16);
      __builtin_memcpy(&b, &blo, 16);
      __builtin_memcpy((char*)&b + 16, &bhi, 16);
    }

    const __bf16* abase = &Abf[kt & 1][0];
#pragma unroll
    for (int s = 0; s < 4; ++s) {
      // A fragment rows 16s+rc, K runs {8hi..8hi+7} and {16+8hi..23+8hi}
      const __bf16* ap = abase + ((s << 4) + rc) * 32 + (hi << 3);
      f32x4 alo = *(const f32x4*)ap;          // 2x ds_load_b128, zero VALU
      f32x4 ahi = *(const f32x4*)(ap + 16);
      v16bf a;
      __builtin_memcpy(&a, &alo, 16);
      __builtin_memcpy((char*)&a + 16, &ahi, 16);
      acc[s] = __builtin_amdgcn_wmma_f32_16x16x32_bf16(
          false, a, false, b, (short)0, acc[s], false, false);
    }

    asm volatile("s_wait_asynccnt 0x0" ::: "memory");  // next tile landed
    __syncthreads();                                   // all waves done w/ buf
  }

  // D layout: VGPR v holds row (v + 8*hi) of its 16x16 tile, col = wave*16+rc
  const int n = (wave << 4) + rc;
  const float bn = HAS_BIAS ? bias[n] : 0.0f;
  if (m0 + 64 <= M) {                                  // full-tile fast path
#pragma unroll
    for (int s = 0; s < 4; ++s)
#pragma unroll
      for (int v = 0; v < 8; ++v) {
        float o = acc[s][v] + bn;
        if (RELU_OUT) o = o > 0.0f ? o : 0.0f;
        C[(size_t)(m0 + (s << 4) + v + 8 * hi) * 128 + n] = o;
      }
  } else {
#pragma unroll
    for (int s = 0; s < 4; ++s)
#pragma unroll
      for (int v = 0; v < 8; ++v) {
        int m = m0 + (s << 4) + v + 8 * hi;
        if (m < M) {
          float o = acc[s][v] + bn;
          if (RELU_OUT) o = o > 0.0f ? o : 0.0f;
          C[(size_t)m * 128 + n] = o;
        }
      }
  }
}

// alpha[i] = dot(H[i, 0:128], a[0:128]); one wave32 per row, float4 per lane
__global__ __launch_bounds__(256)
void rowdot(const float* __restrict__ H, const float* __restrict__ a,
            float* __restrict__ out, int N) {
  int row  = blockIdx.x * 8 + (threadIdx.x >> 5);
  int lane = threadIdx.x & 31;
  if (row >= N) return;
  f32x4 hv = ((const f32x4*)(H + (size_t)row * 128))[lane];
  f32x4 av = ((const f32x4*)a)[lane];
  float s = hv.x * av.x + hv.y * av.y + hv.z * av.z + hv.w * av.w;
#pragma unroll
  for (int m = 16; m > 0; m >>= 1) s += __shfl_xor(s, m, 32);
  if (lane == 0) out[row] = s;
}

__global__ void seg_init(unsigned* __restrict__ amax, float* __restrict__ denom, int N) {
  int i = blockIdx.x * 256 + threadIdx.x;
  if (i < N) { amax[i] = ENC_NEG_INF; denom[i] = 0.0f; }
}

__global__ void zero_f(float* __restrict__ p, size_t n) {
  size_t i = (size_t)blockIdx.x * 256 + threadIdx.x;
  if (i < n) p[i] = 0.0f;
}

// pass 1: segment max of leaky_relu(alpha_src[s] + alpha_dst[d]) over d
__global__ void edge_max(const int* __restrict__ src, const int* __restrict__ dst,
                         const float* __restrict__ as, const float* __restrict__ ad,
                         unsigned* __restrict__ amax, int E) {
  int e = blockIdx.x * 256 + threadIdx.x;
  if (e >= E) return;
  int d = dst[e];
  float al = as[src[e]] + ad[d];
  al = al > 0.0f ? al : 0.2f * al;
  atomicMax(&amax[d], fenc(al));
}

// pass 2: ex = exp(alpha - amax[d]); denom[d] += ex
__global__ void edge_exp(const int* __restrict__ src, const int* __restrict__ dst,
                         const float* __restrict__ as, const float* __restrict__ ad,
                         const unsigned* __restrict__ amax, float* __restrict__ denom,
                         float* __restrict__ ex, int E) {
  int e = blockIdx.x * 256 + threadIdx.x;
  if (e >= E) return;
  int d = dst[e];
  float al = as[src[e]] + ad[d];
  al = al > 0.0f ? al : 0.2f * al;
  float x = expf(al - fdec(amax[d]));
  ex[e] = x;
  atomicAdd(&denom[d], x);
}

// pass 3: out[d] += hs[s] * (ex / (denom[d] + 1e-16)); one wave32 per edge
__global__ __launch_bounds__(256)
void edge_scatter(const int* __restrict__ src, const int* __restrict__ dst,
                  const float* __restrict__ hs, const float* __restrict__ denom,
                  const float* __restrict__ ex, float* __restrict__ out, int E) {
  int e    = blockIdx.x * 8 + (threadIdx.x >> 5);
  int lane = threadIdx.x & 31;
  if (e >= E) return;
  int s = src[e], d = dst[e];
  float coef = ex[e] / (denom[d] + 1e-16f);
  f32x4 h = ((const f32x4*)(hs + (size_t)s * 128))[lane];
  float* o = out + (size_t)d * 128 + lane * 4;
  atomicAdd(o + 0, h.x * coef);
  atomicAdd(o + 1, h.y * coef);
  atomicAdd(o + 2, h.z * coef);
  atomicAdd(o + 3, h.w * coef);
}

// out[i, :] += b[:]  (GATConv bias, added to every destination node)
__global__ void add_bias(float* __restrict__ out, const float* __restrict__ b, int N) {
  int i = blockIdx.x * 256 + threadIdx.x;
  if (i < N * 128) out[i] += b[i & 127];
}

// ---------------------------------------------------------------------------
extern "C" void kernel_launch(void* const* d_in, const int* in_sizes, int n_in,
                              void* d_out, int out_size, void* d_ws, size_t ws_size,
                              hipStream_t stream) {
  (void)in_sizes; (void)n_in; (void)out_size; (void)ws_size;

  const float* x_drug = (const float*)d_in[0];
  const float* x_dis  = (const float*)d_in[1];
  const int*   edges[7];
  for (int r = 0; r < 7; ++r) edges[r] = (const int*)d_in[2 + r];
  const float* W1s = (const float*)d_in[9];
  const float* W1d = (const float*)d_in[10];
  const float* a1s = (const float*)d_in[11];
  const float* a1d = (const float*)d_in[12];
  const float* b1  = (const float*)d_in[13];
  const float* W2s = (const float*)d_in[14];
  const float* W2d = (const float*)d_in[15];
  const float* a2s = (const float*)d_in[16];
  const float* a2d = (const float*)d_in[17];
  const float* b2  = (const float*)d_in[18];
  const float* W_drug = (const float*)d_in[19];
  const float* b_drug = (const float*)d_in[20];
  const float* W_dis  = (const float*)d_in[21];
  const float* b_dis  = (const float*)d_in[22];

  float* ws = (float*)d_ws;
  float* xd1 = ws;                 ws += (size_t)NDRUG * HIDC;
  float* xi1 = ws;                 ws += (size_t)NDIS  * HIDC;
  float* xd2 = ws;                 ws += (size_t)NDRUG * HIDC;
  float* xi2 = ws;                 ws += (size_t)NDIS  * HIDC;
  float* hs  = ws;                 ws += (size_t)NDIS  * HIDC;
  float* hd  = ws;                 ws += (size_t)NDIS  * HIDC;
  float* asrc = ws;                ws += NDIS;
  float* adst = ws;                ws += NDIS;
  unsigned* amax = (unsigned*)ws;  ws += NDIS;
  float* denom = ws;               ws += NDIS;
  float* exbuf = ws;               ws += NEDGE;
  __bf16* bpack = (__bf16*)ws;     ws += (DIN * HIDC) / 2 + 16;  // 256*128 bf16
  __bf16* xb0 = (__bf16*)ws;       ws += ((size_t)NDRUG * DIN) / 2 + 16;
  __bf16* xb1 = (__bf16*)ws;       ws += ((size_t)NDIS  * DIN) / 2 + 16;

  static const int SRCt[7] = {0, 1, 0, 1, 1, 1, 0};
  static const int DSTt[7] = {1, 0, 1, 0, 1, 1, 0};
  const int nsz[2] = {NDRUG, NDIS};

  auto cvt = [&](const float* X, __bf16* Y, size_t n, int relu) {
    cvt_bf16<<<(unsigned)((n / 8 + 255) / 256), 256, 0, stream>>>(X, Y, n, relu);
  };
  auto run_gemm = [&](const __bf16* X, const float* W, float* Y, int M, int Kin) {
    pack_B<<<(Kin * 128 + 255) / 256, 256, 0, stream>>>(W, bpack, Kin);
    gemm_wmma<false, false><<<(M + 63) / 64, 256, 0, stream>>>(X, bpack, nullptr,
                                                               Y, M, Kin);
  };

  auto run_layer = [&](const float* xin0, const float* xin1, int Kin, int relu_in,
                       const float* WsAll, const float* WdAll,
                       const float* asAll, const float* adAll, const float* bAll,
                       float* out0, float* out1) {
    // bf16 versions of both feature matrices, reused by all 14 GEMMs
    cvt(xin0, xb0, (size_t)NDRUG * Kin, relu_in);
    cvt(xin1, xb1, (size_t)NDIS * Kin, relu_in);
    zero_f<<<(unsigned)(((size_t)NDRUG * HIDC + 255) / 256), 256, 0, stream>>>(
        out0, (size_t)NDRUG * HIDC);
    zero_f<<<(unsigned)(((size_t)NDIS * HIDC + 255) / 256), 256, 0, stream>>>(
        out1, (size_t)NDIS * HIDC);
    const __bf16* xb[2] = {xb0, xb1};
    float* outp[2] = {out0, out1};
    for (int r = 0; r < 7; ++r) {
      const int Ns = nsz[SRCt[r]], Nd = nsz[DSTt[r]];
      const int* src = edges[r];
      const int* dst = edges[r] + NEDGE;
      run_gemm(xb[SRCt[r]], WsAll + (size_t)r * Kin * HIDC, hs, Ns, Kin);
      run_gemm(xb[DSTt[r]], WdAll + (size_t)r * Kin * HIDC, hd, Nd, Kin);
      rowdot<<<(Ns + 7) / 8, 256, 0, stream>>>(hs, asAll + r * HIDC, asrc, Ns);
      rowdot<<<(Nd + 7) / 8, 256, 0, stream>>>(hd, adAll + r * HIDC, adst, Nd);
      seg_init<<<(Nd + 255) / 256, 256, 0, stream>>>(amax, denom, Nd);
      edge_max<<<(NEDGE + 255) / 256, 256, 0, stream>>>(src, dst, asrc, adst, amax, NEDGE);
      edge_exp<<<(NEDGE + 255) / 256, 256, 0, stream>>>(src, dst, asrc, adst, amax,
                                                        denom, exbuf, NEDGE);
      edge_scatter<<<(NEDGE + 7) / 8, 256, 0, stream>>>(src, dst, hs, denom, exbuf,
                                                        outp[DSTt[r]], NEDGE);
      add_bias<<<(Nd * HIDC + 255) / 256, 256, 0, stream>>>(outp[DSTt[r]],
                                                            bAll + r * HIDC, Nd);
    }
  };

  // layer 1 (K=256), layer 2 (K=128, inter-layer ReLU fused into cvt pass)
  run_layer(x_drug, x_dis, DIN, 0, W1s, W1d, a1s, a1d, b1, xd1, xi1);
  run_layer(xd1, xi1, HIDC, 1, W2s, W2d, a2s, a2d, b2, xd2, xi2);

  // output projections with fused bias + ReLU, written straight to d_out
  float* outD = (float*)d_out;
  float* outI = outD + (size_t)NDRUG * HIDC;
  cvt(xd2, xb0, (size_t)NDRUG * HIDC, 0);
  cvt(xi2, xb1, (size_t)NDIS * HIDC, 0);
  pack_B<<<(HIDC * 128 + 255) / 256, 256, 0, stream>>>(W_drug, bpack, HIDC);
  gemm_wmma<true, true><<<(NDRUG + 63) / 64, 256, 0, stream>>>(xb0, bpack, b_drug,
                                                               outD, NDRUG, HIDC);
  pack_B<<<(HIDC * 128 + 255) / 256, 256, 0, stream>>>(W_dis, bpack, HIDC);
  gemm_wmma<true, true><<<(NDIS + 63) / 64, 256, 0, stream>>>(xb1, bpack, b_dis,
                                                              outI, NDIS, HIDC);
}
